// DNC_4243427688821
// MI455X (gfx1250) — matrix operand order repaired and checked by reading, not compiled, not run on hardware
//
#include <hip/hip_runtime.h>
#include <math.h>

#define BB 64
#define LL 256
#define EE 256
#define HH 512
#define MM 256
#define HD 64
#define RR 4
#define OO 1024
#define CI 512
#define TO 768
#define G4H 2048
#define KLSTM 1024
#define LN_EPS 1e-5f

typedef __attribute__((ext_vector_type(16))) __bf16 bf16x16;
typedef __attribute__((ext_vector_type(8)))  float  f32x8;

union Frag {
  bf16x16 v;
  uint4 q[2];
  unsigned short u[16];
};

__device__ __forceinline__ unsigned short f2bf(float f) {
  unsigned int u = __float_as_uint(f);
  unsigned int r = u + 0x7FFFu + ((u >> 16) & 1u);   // round-to-nearest-even
  return (unsigned short)(r >> 16);
}
__device__ __forceinline__ float sigmoidf_(float x) { return 1.f / (1.f + __expf(-x)); }
__device__ __forceinline__ float wsum(float x) {
#pragma unroll
  for (int o = 16; o > 0; o >>= 1) x += __shfl_xor(x, o, 32);
  return x;
}
__device__ __forceinline__ float wmax(float x) {
#pragma unroll
  for (int o = 16; o > 0; o >>= 1) x = fmaxf(x, __shfl_xor(x, o, 32));
  return x;
}

// ---------------------------------------------------------------------------
// Weight prep: transpose to K-major and convert to bf16; fuse biases.
//   Wt_lstm [1024][2048]: rows 0..511 = W_ih^T, rows 512..1023 = W_hh^T
//   Wt_ctrl [512][512]:   cols 0..255 rk | 256..319 wk | 320..383 er |
//                         384..447 ad | 448 ws | 449..511 zero
//   Wt_projT[768][1024] = W_proj^T
// ---------------------------------------------------------------------------
__global__ void dnc_prep(
    const float* __restrict__ W_ih, const float* __restrict__ W_hh,
    const float* __restrict__ b_ih, const float* __restrict__ b_hh,
    const float* __restrict__ W_rk, const float* __restrict__ b_rk,
    const float* __restrict__ W_wk, const float* __restrict__ b_wk,
    const float* __restrict__ W_ws, const float* __restrict__ b_ws,
    const float* __restrict__ W_er, const float* __restrict__ b_er,
    const float* __restrict__ W_ad, const float* __restrict__ b_ad,
    const float* __restrict__ W_proj,
    unsigned short* __restrict__ Wt_lstm,
    unsigned short* __restrict__ Wt_ctrl,
    unsigned short* __restrict__ Wt_projT,
    float* __restrict__ b_lstm, float* __restrict__ b_ctrl)
{
  const size_t stride = (size_t)gridDim.x * blockDim.x;
  const size_t t0 = (size_t)blockIdx.x * blockDim.x + threadIdx.x;

  for (size_t i = t0; i < (size_t)KLSTM * G4H; i += stride) {
    int k = (int)(i / G4H), j = (int)(i % G4H);
    float w = (k < CI) ? W_ih[(size_t)j * CI + k] : W_hh[(size_t)j * HH + (k - CI)];
    Wt_lstm[i] = f2bf(w);
  }
  for (size_t i = t0; i < (size_t)CI * CI; i += stride) {
    int k = (int)(i / CI), n = (int)(i % CI);
    float w;
    if      (n < 256) w = W_rk[(size_t)n * HH + k];
    else if (n < 320) w = W_wk[(size_t)(n - 256) * HH + k];
    else if (n < 384) w = W_er[(size_t)(n - 320) * HH + k];
    else if (n < 448) w = W_ad[(size_t)(n - 384) * HH + k];
    else if (n == 448) w = W_ws[k];
    else w = 0.f;
    Wt_ctrl[i] = f2bf(w);
  }
  for (size_t i = t0; i < (size_t)TO * OO; i += stride) {
    int k = (int)(i / OO), j = (int)(i % OO);
    Wt_projT[i] = f2bf(W_proj[(size_t)j * TO + k]);
  }
  for (size_t i = t0; i < G4H; i += stride) b_lstm[i] = b_ih[i] + b_hh[i];
  for (size_t i = t0; i < CI; i += stride) {
    int n = (int)i; float b;
    if      (n < 256) b = b_rk[n];
    else if (n < 320) b = b_wk[n - 256];
    else if (n < 384) b = b_er[n - 320];
    else if (n < 448) b = b_ad[n - 384];
    else if (n == 448) b = b_ws[0];
    else b = 0.f;
    b_ctrl[n] = b;
  }
}

__global__ void dnc_init_mem(float* __restrict__ mem, int n) {
  int stride = gridDim.x * blockDim.x;
  for (int i = blockIdx.x * blockDim.x + threadIdx.x; i < n; i += stride) mem[i] = 0.f;
}

// ---------------------------------------------------------------------------
// Persistent per-batch-group kernel: 1 block = 16 batch rows, full t-loop.
// ---------------------------------------------------------------------------
__global__ __launch_bounds__(512) void dnc_main(
    const float* __restrict__ x_emb,
    const float* __restrict__ ln_in_g,  const float* __restrict__ ln_in_b,
    const float* __restrict__ ln_h_g,   const float* __restrict__ ln_h_b,
    const float* __restrict__ ln_rk_g,  const float* __restrict__ ln_rk_b,
    const float* __restrict__ ln_wk_g,  const float* __restrict__ ln_wk_b,
    const float* __restrict__ ln_mem_g, const float* __restrict__ ln_mem_b,
    const float* __restrict__ ln_out_g, const float* __restrict__ ln_out_b,
    const float* __restrict__ b_proj,
    const unsigned short* __restrict__ Wt_lstm,
    const unsigned short* __restrict__ Wt_ctrl,
    const unsigned short* __restrict__ Wt_projT,
    const float* __restrict__ b_lstm, const float* __restrict__ b_ctrl,
    float* __restrict__ out_logits,   // [B][L][O]
    float* __restrict__ mem,          // [B][M][HD] working + final memory
    float* __restrict__ out_h,        // [B][H]
    float* __restrict__ out_c)        // [B][H]
{
  __shared__ float sh_h [16][HH];
  __shared__ float sh_c [16][HH];
  __shared__ float sh_hn[16][HH];
  __shared__ __align__(16) unsigned short sh_hnb[16][HH];
  __shared__ float sh_rv[16][RR * HD];
  __shared__ float sh_rk[16][RR][HD];
  __shared__ float sh_wk[16][HD];
  __shared__ float sh_ew[16][2 * HD];   // [0..63]=erase, [64..127]=write_vec
  __shared__ float sh_ws[16];
  union U1 {                            // phase-disjoint overlays
    unsigned short a_lstm[16][KLSTM];   // phases A,B
    float          wscore[16][MM];      // phase E
    unsigned short outb[16][TO];        // phase G
  };
  union U2 {
    float ctrl[16][CI];                 // phases D,E
    float rscore[16][RR][MM];           // phase F
  };
  __shared__ __align__(16) U1 u1;
  __shared__ __align__(16) U2 u2;

  const int tid  = threadIdx.x;
  const int wid  = tid >> 5;       // 0..15  (wave per batch row for row phases)
  const int lane = tid & 31;
  const int grp  = lane >> 4;      // lane group for WMMA A/C layout
  const int Mr   = lane & 15;
  const int b0   = blockIdx.x * 16;

  for (int i = tid; i < 16 * HH; i += 512) { (&sh_h[0][0])[i] = 0.f; (&sh_c[0][0])[i] = 0.f; }
  for (int i = tid; i < 16 * RR * HD; i += 512) (&sh_rv[0][0])[i] = 0.f;
  __syncthreads();

  for (int t = 0; t < LL; ++t) {
    // ---- Phase A: ci = LN(concat(x_t, read_vec)); stage [ci | h] as bf16 ---
    {
      const int row = wid;
      float x[16];
#pragma unroll
      for (int i = 0; i < 16; ++i) {
        int k = i * 32 + lane;
        x[i] = (i < 8) ? x_emb[((size_t)(b0 + row) * LL + t) * EE + k]
                       : sh_rv[row][k - 256];
      }
      float s = 0.f;
#pragma unroll
      for (int i = 0; i < 16; ++i) s += x[i];
      float mu = wsum(s) * (1.f / 512.f);
      float v = 0.f;
#pragma unroll
      for (int i = 0; i < 16; ++i) { float d = x[i] - mu; v += d * d; }
      float rstd = rsqrtf(wsum(v) * (1.f / 512.f) + LN_EPS);
#pragma unroll
      for (int i = 0; i < 16; ++i) {
        int k = i * 32 + lane;
        u1.a_lstm[row][k]      = f2bf((x[i] - mu) * rstd * ln_in_g[k] + ln_in_b[k]);
        u1.a_lstm[row][CI + k] = f2bf(sh_h[row][k]);
      }
    }
    __syncthreads();

    // ---- Phase B: gates = [ci|h]·Wt_lstm (WMMA bf16), fused LSTM update ----
#pragma unroll
    for (int half = 0; half < 2; ++half) {
      const int n0 = (wid + 16 * half) * 16;        // column base within H
      f32x8 acc0 = {0,0,0,0,0,0,0,0}, acc1 = acc0, acc2 = acc0, acc3 = acc0;
      for (int kb = 0; kb < KLSTM; kb += 32) {
        Frag A;
        const unsigned short* ap = &u1.a_lstm[Mr][kb + 8 * grp];
        A.q[0] = *(const uint4*)ap;
        A.q[1] = *(const uint4*)(ap + 16);
        const unsigned short* bp = Wt_lstm + (size_t)(kb + 8 * grp) * G4H + n0 + Mr;
        Frag B0, B1, B2, B3;
#pragma unroll
        for (int s = 0; s < 16; ++s) {
          size_t o = (size_t)((s & 7) + ((s >> 3) << 4)) * G4H;
          B0.u[s] = bp[o]; B1.u[s] = bp[o + 512]; B2.u[s] = bp[o + 1024]; B3.u[s] = bp[o + 1536];
        }
        acc0 = __builtin_amdgcn_wmma_f32_16x16x32_bf16(false, A.v, false, B0.v, (short)0, acc0, false, false);
        acc1 = __builtin_amdgcn_wmma_f32_16x16x32_bf16(false, A.v, false, B1.v, (short)0, acc1, false, false);
        acc2 = __builtin_amdgcn_wmma_f32_16x16x32_bf16(false, A.v, false, B2.v, (short)0, acc2, false, false);
        acc3 = __builtin_amdgcn_wmma_f32_16x16x32_bf16(false, A.v, false, B3.v, (short)0, acc3, false, false);
      }
      const int col = n0 + Mr;
      const float bi  = b_lstm[col];
      const float bf_ = b_lstm[col + HH];
      const float bg_ = b_lstm[col + 2 * HH];
      const float bo  = b_lstm[col + 3 * HH];
#pragma unroll
      for (int r = 0; r < 8; ++r) {
        int mrow = r + 8 * grp;
        float iv = sigmoidf_(acc0[r] + bi);
        float fv = sigmoidf_(acc1[r] + bf_);
        float gv = tanhf(acc2[r] + bg_);
        float ov = sigmoidf_(acc3[r] + bo);
        float cn = fv * sh_c[mrow][col] + iv * gv;
        sh_c[mrow][col] = cn;
        sh_h[mrow][col] = ov * tanhf(cn);
      }
    }
    __syncthreads();

    // ---- Phase C: hn = LN(h) ----------------------------------------------
    {
      const int row = wid;
      float x[16];
#pragma unroll
      for (int i = 0; i < 16; ++i) x[i] = sh_h[row][i * 32 + lane];
      float s = 0.f;
#pragma unroll
      for (int i = 0; i < 16; ++i) s += x[i];
      float mu = wsum(s) * (1.f / 512.f);
      float v = 0.f;
#pragma unroll
      for (int i = 0; i < 16; ++i) { float d = x[i] - mu; v += d * d; }
      float rstd = rsqrtf(wsum(v) * (1.f / 512.f) + LN_EPS);
#pragma unroll
      for (int i = 0; i < 16; ++i) {
        int k = i * 32 + lane;
        float y = (x[i] - mu) * rstd * ln_h_g[k] + ln_h_b[k];
        sh_hn[row][k]  = y;
        sh_hnb[row][k] = f2bf(y);
      }
    }
    __syncthreads();

    // ---- Phase D: controller heads  hn·Wt_ctrl (WMMA) ---------------------
#pragma unroll
    for (int half = 0; half < 2; ++half) {
      const int n0 = (wid + 16 * half) * 16;
      f32x8 acc = {0,0,0,0,0,0,0,0};
      for (int kb = 0; kb < CI; kb += 32) {
        Frag A;
        const unsigned short* ap = &sh_hnb[Mr][kb + 8 * grp];
        A.q[0] = *(const uint4*)ap;
        A.q[1] = *(const uint4*)(ap + 16);
        Frag Bf;
        const unsigned short* bp = Wt_ctrl + (size_t)(kb + 8 * grp) * CI + n0 + Mr;
#pragma unroll
        for (int s = 0; s < 16; ++s) Bf.u[s] = bp[(size_t)((s & 7) + ((s >> 3) << 4)) * CI];
        acc = __builtin_amdgcn_wmma_f32_16x16x32_bf16(false, A.v, false, Bf.v, (short)0, acc, false, false);
      }
      const int col = n0 + Mr;
      const float bb = b_ctrl[col];
#pragma unroll
      for (int r = 0; r < 8; ++r) u2.ctrl[r + 8 * grp][col] = acc[r] + bb;
    }
    __syncthreads();

    // ---- Phase E1: LN write/read keys, erase/write vectors, write_str -----
    {
      const int row = wid;
      {
        float x0 = u2.ctrl[row][256 + lane];
        float x1 = u2.ctrl[row][256 + 32 + lane];
        float mu = wsum(x0 + x1) * (1.f / 64.f);
        float d0 = x0 - mu, d1 = x1 - mu;
        float rstd = rsqrtf(wsum(d0 * d0 + d1 * d1) * (1.f / 64.f) + LN_EPS);
        sh_wk[row][lane]      = d0 * rstd * ln_wk_g[lane]      + ln_wk_b[lane];
        sh_wk[row][lane + 32] = d1 * rstd * ln_wk_g[lane + 32] + ln_wk_b[lane + 32];
      }
#pragma unroll
      for (int seg = 0; seg < RR; ++seg) {
        float x0 = u2.ctrl[row][seg * 64 + lane];
        float x1 = u2.ctrl[row][seg * 64 + 32 + lane];
        float mu = wsum(x0 + x1) * (1.f / 64.f);
        float d0 = x0 - mu, d1 = x1 - mu;
        float rstd = rsqrtf(wsum(d0 * d0 + d1 * d1) * (1.f / 64.f) + LN_EPS);
        sh_rk[row][seg][lane]      = d0 * rstd * ln_rk_g[lane]      + ln_rk_b[lane];
        sh_rk[row][seg][lane + 32] = d1 * rstd * ln_rk_g[lane + 32] + ln_rk_b[lane + 32];
      }
      sh_ew[row][lane]           = sigmoidf_(u2.ctrl[row][320 + lane]);
      sh_ew[row][lane + 32]      = sigmoidf_(u2.ctrl[row][320 + 32 + lane]);
      sh_ew[row][64 + lane]      = tanhf(u2.ctrl[row][384 + lane]);
      sh_ew[row][64 + lane + 32] = tanhf(u2.ctrl[row][384 + 32 + lane]);
      if (lane == 0) sh_ws[row] = sigmoidf_(u2.ctrl[row][448]);
    }
    __syncthreads();

    // ---- Phase E2: write scores = <LN(mem_m), wk> -------------------------
#pragma unroll 1
    for (int p = 0; p < 8; ++p) {
      int idx = tid + p * 512, row = idx >> 8, m = idx & 255;
      const float* mrow = mem + ((size_t)(b0 + row) * MM + m) * HD;
      float s = 0.f, ss = 0.f;
      for (int j = 0; j < HD; ++j) { float x = mrow[j]; s += x; ss += x * x; }
      float mu = s * (1.f / 64.f);
      float rstd = rsqrtf(fmaxf(ss * (1.f / 64.f) - mu * mu, 0.f) + LN_EPS);
      float sc = 0.f;
      for (int j = 0; j < HD; ++j) {
        float xn = (mrow[j] - mu) * rstd * ln_mem_g[j] + ln_mem_b[j];
        sc += xn * sh_wk[row][j];
      }
      u1.wscore[row][m] = sc;
    }
    __syncthreads();

    // ---- Phase E3: softmax over M, scale by write_str ---------------------
    {
      const int row = wid;
      float v[8];
#pragma unroll
      for (int i = 0; i < 8; ++i) v[i] = u1.wscore[row][i * 32 + lane];
      float mx = v[0];
#pragma unroll
      for (int i = 1; i < 8; ++i) mx = fmaxf(mx, v[i]);
      mx = wmax(mx);
      float s = 0.f;
#pragma unroll
      for (int i = 0; i < 8; ++i) { v[i] = __expf(v[i] - mx); s += v[i]; }
      s = wsum(s);
      float scale = sh_ws[row] / s;
#pragma unroll
      for (int i = 0; i < 8; ++i) u1.wscore[row][i * 32 + lane] = v[i] * scale;
    }
    __syncthreads();

    // ---- Phase E4: memory erase/write -------------------------------------
#pragma unroll 1
    for (int p = 0; p < 8; ++p) {
      int idx = tid + p * 512, row = idx >> 8, m = idx & 255;
      float w = u1.wscore[row][m];
      float* mrow = mem + ((size_t)(b0 + row) * MM + m) * HD;
      for (int j = 0; j < HD; ++j) {
        float er = sh_ew[row][j], wv = sh_ew[row][HD + j];
        mrow[j] = mrow[j] * (1.f - w * er) + w * wv;
      }
    }
    __threadfence();
    __syncthreads();

    // ---- Phase F1: read scores = <LN(mem'_m), rk_k> -----------------------
#pragma unroll 1
    for (int p = 0; p < 8; ++p) {
      int idx = tid + p * 512, row = idx >> 8, m = idx & 255;
      const float* mrow = mem + ((size_t)(b0 + row) * MM + m) * HD;
      float s = 0.f, ss = 0.f;
      for (int j = 0; j < HD; ++j) { float x = mrow[j]; s += x; ss += x * x; }
      float mu = s * (1.f / 64.f);
      float rstd = rsqrtf(fmaxf(ss * (1.f / 64.f) - mu * mu, 0.f) + LN_EPS);
      float a0 = 0.f, a1 = 0.f, a2 = 0.f, a3 = 0.f;
      for (int j = 0; j < HD; ++j) {
        float xn = (mrow[j] - mu) * rstd * ln_mem_g[j] + ln_mem_b[j];
        a0 += xn * sh_rk[row][0][j];
        a1 += xn * sh_rk[row][1][j];
        a2 += xn * sh_rk[row][2][j];
        a3 += xn * sh_rk[row][3][j];
      }
      u2.rscore[row][0][m] = a0;
      u2.rscore[row][1][m] = a1;
      u2.rscore[row][2][m] = a2;
      u2.rscore[row][3][m] = a3;
    }
    __syncthreads();

    // ---- Phase F2: per-key softmax over M ---------------------------------
    {
      const int row = wid;
      for (int k = 0; k < RR; ++k) {
        float v[8];
#pragma unroll
        for (int i = 0; i < 8; ++i) v[i] = u2.rscore[row][k][i * 32 + lane];
        float mx = v[0];
#pragma unroll
        for (int i = 1; i < 8; ++i) mx = fmaxf(mx, v[i]);
        mx = wmax(mx);
        float s = 0.f;
#pragma unroll
        for (int i = 0; i < 8; ++i) { v[i] = __expf(v[i] - mx); s += v[i]; }
        float inv = 1.f / wsum(s);
#pragma unroll
        for (int i = 0; i < 8; ++i) u2.rscore[row][k][i * 32 + lane] = v[i] * inv;
      }
    }
    __syncthreads();

    // ---- Phase F3: read_vec = rw · memory ---------------------------------
#pragma unroll 1
    for (int p = 0; p < 8; ++p) {
      int idx = tid + p * 512, row = idx >> 8, col = idx & 255;
      int k = col >> 6, jj = col & 63;
      const float* mb = mem + (size_t)(b0 + row) * MM * HD + jj;
      const float* rw = &u2.rscore[row][k][0];
      float acc = 0.f;
      for (int m = 0; m < MM; ++m) acc += rw[m] * mb[(size_t)m * HD];
      sh_rv[row][col] = acc;
    }
    __syncthreads();

    // ---- Phase G1: out = LN(concat(hn, read_vec)) -> bf16 -----------------
    {
      const int row = wid;
      float x[24];
#pragma unroll
      for (int i = 0; i < 24; ++i) {
        int k = i * 32 + lane;
        x[i] = (i < 16) ? sh_hn[row][k] : sh_rv[row][k - 512];
      }
      float s = 0.f;
#pragma unroll
      for (int i = 0; i < 24; ++i) s += x[i];
      float mu = wsum(s) * (1.f / 768.f);
      float v = 0.f;
#pragma unroll
      for (int i = 0; i < 24; ++i) { float d = x[i] - mu; v += d * d; }
      float rstd = rsqrtf(wsum(v) * (1.f / 768.f) + LN_EPS);
#pragma unroll
      for (int i = 0; i < 24; ++i) {
        int k = i * 32 + lane;
        u1.outb[row][k] = f2bf((x[i] - mu) * rstd * ln_out_g[k] + ln_out_b[k]);
      }
    }
    __syncthreads();

    // ---- Phase G2: logits = out·Wt_proj (WMMA) -> global ------------------
#pragma unroll
    for (int q = 0; q < 4; ++q) {
      const int n0 = (wid + 16 * q) * 16;
      f32x8 acc = {0,0,0,0,0,0,0,0};
      for (int kb = 0; kb < TO; kb += 32) {
        Frag A;
        const unsigned short* ap = &u1.outb[Mr][kb + 8 * grp];
        A.q[0] = *(const uint4*)ap;
        A.q[1] = *(const uint4*)(ap + 16);
        Frag Bf;
        const unsigned short* bp = Wt_projT + (size_t)(kb + 8 * grp) * OO + n0 + Mr;
#pragma unroll
        for (int s = 0; s < 16; ++s) Bf.u[s] = bp[(size_t)((s & 7) + ((s >> 3) << 4)) * OO];
        acc = __builtin_amdgcn_wmma_f32_16x16x32_bf16(false, A.v, false, Bf.v, (short)0, acc, false, false);
      }
      const int col = n0 + Mr;
      const float bb = b_proj[col];
#pragma unroll
      for (int r = 0; r < 8; ++r) {
        int mrow = r + 8 * grp;
        out_logits[((size_t)(b0 + mrow) * LL + t) * OO + col] = acc[r] + bb;
      }
    }
    __syncthreads();
  }

  // final h, c
  for (int i = tid; i < 16 * HH; i += 512) {
    int row = i >> 9, j = i & 511;
    out_h[(size_t)(b0 + row) * HH + j] = sh_h[row][j];
    out_c[(size_t)(b0 + row) * HH + j] = sh_c[row][j];
  }
}

// ---------------------------------------------------------------------------
extern "C" void kernel_launch(void* const* d_in, const int* in_sizes, int n_in,
                              void* d_out, int out_size, void* d_ws, size_t ws_size,
                              hipStream_t stream)
{
  (void)in_sizes; (void)n_in; (void)out_size; (void)ws_size;

  const float* x_emb    = (const float*)d_in[0];
  const float* W_ih     = (const float*)d_in[1];
  const float* W_hh     = (const float*)d_in[2];
  const float* b_ih     = (const float*)d_in[3];
  const float* b_hh     = (const float*)d_in[4];
  const float* ln_in_g  = (const float*)d_in[5];
  const float* ln_in_b  = (const float*)d_in[6];
  const float* ln_h_g   = (const float*)d_in[7];
  const float* ln_h_b   = (const float*)d_in[8];
  const float* W_rk     = (const float*)d_in[9];
  const float* b_rk     = (const float*)d_in[10];
  const float* W_wk     = (const float*)d_in[11];
  const float* b_wk     = (const float*)d_in[12];
  const float* W_ws     = (const float*)d_in[13];
  const float* b_ws     = (const float*)d_in[14];
  const float* W_er     = (const float*)d_in[15];
  const float* b_er     = (const float*)d_in[16];
  const float* W_ad     = (const float*)d_in[17];
  const float* b_ad     = (const float*)d_in[18];
  const float* ln_rk_g  = (const float*)d_in[19];
  const float* ln_rk_b  = (const float*)d_in[20];
  const float* ln_wk_g  = (const float*)d_in[21];
  const float* ln_wk_b  = (const float*)d_in[22];
  const float* ln_mem_g = (const float*)d_in[23];
  const float* ln_mem_b = (const float*)d_in[24];
  const float* ln_out_g = (const float*)d_in[25];
  const float* ln_out_b = (const float*)d_in[26];
  const float* W_proj   = (const float*)d_in[27];
  const float* b_proj   = (const float*)d_in[28];

  float* out        = (float*)d_out;
  float* out_logits = out;                               // [64][256][1024]
  float* mem        = out + (size_t)BB * LL * OO;        // [64][256][64]
  float* out_h      = mem + (size_t)BB * MM * HD;        // [64][512]
  float* out_c      = out_h + (size_t)BB * HH;           // [64][512]

  char* ws = (char*)d_ws;
  unsigned short* Wt_lstm  = (unsigned short*)ws;                                   // 4 MiB
  unsigned short* Wt_ctrl  = (unsigned short*)(ws + (size_t)4 * 1024 * 1024);       // 512 KiB
  unsigned short* Wt_projT = (unsigned short*)(ws + (size_t)4 * 1024 * 1024 + 512 * 1024); // 1.5 MiB
  float* b_lstm = (float*)(ws + (size_t)4 * 1024 * 1024 + 512 * 1024 + 1536 * 1024);
  float* b_ctrl = b_lstm + G4H;

  dnc_prep<<<2048, 256, 0, stream>>>(W_ih, W_hh, b_ih, b_hh, W_rk, b_rk, W_wk, b_wk,
                                     W_ws, b_ws, W_er, b_er, W_ad, b_ad, W_proj,
                                     Wt_lstm, Wt_ctrl, Wt_projT, b_lstm, b_ctrl);
  dnc_init_mem<<<512, 256, 0, stream>>>(mem, BB * MM * HD);
  dnc_main<<<BB / 16, 512, 0, stream>>>(x_emb,
      ln_in_g, ln_in_b, ln_h_g, ln_h_b, ln_rk_g, ln_rk_b, ln_wk_g, ln_wk_b,
      ln_mem_g, ln_mem_b, ln_out_g, ln_out_b, b_proj,
      Wt_lstm, Wt_ctrl, Wt_projT, b_lstm, b_ctrl,
      out_logits, mem, out_h, out_c);
}
